// encoder_52072183497350
// MI455X (gfx1250) — matrix operand order repaired and checked by reading, not compiled, not run on hardware
//
#include <hip/hip_runtime.h>
#include <hip/hip_bf16.h>
#include <math.h>

// ---------------- model constants ----------------
#define HIDDEN   1024
#define HEADS    16
#define DHEAD    64
#define SEQ      1024
#define BATCH    4
#define ROWS     (BATCH * SEQ)   // 4096
#define NBLOCKS  12
#define EPS_LN   1e-3f
// softmax scale folded into Q at projection time, in log2 domain:
// (1/sqrt(HIDDEN)) * log2(e) = (1/32) * 1.4426950408889634
#define QSCALE_LOG2E 0.045084220027780106f

typedef __bf16 bf16_t;
typedef __attribute__((ext_vector_type(16))) __bf16 v16bf;
typedef __attribute__((ext_vector_type(8)))  __bf16 v8bf;
typedef __attribute__((ext_vector_type(8)))  float  v8f;

static __device__ __forceinline__ v8f wmma_bf16(v16bf a, v16bf b, v8f c) {
    // D = A(16x32 bf16) x B(32x16 bf16) + C(16x16 f32)
    return __builtin_amdgcn_wmma_f32_16x16x32_bf16(
        false, a, false, b, (short)0, c, false, false);
}

static __device__ __forceinline__ v16bf pack16(v8bf lo, v8bf hi) {
    v16bf r;
#pragma unroll
    for (int j = 0; j < 8; ++j) { r[j] = lo[j]; r[8 + j] = hi[j]; }
    return r;
}

// ---------------- weight transpose + fp32->bf16 convert ----------------
// Wt[n][k] = (bf16) W[k][n], per 1024x1024 matrix (blockIdx.z selects matrix)
__global__ __launch_bounds__(256)
void transpose_cvt_kernel(const float* __restrict__ W, bf16_t* __restrict__ Wt) {
    __shared__ float tile[32][33];
    const size_t moff = (size_t)blockIdx.z * HIDDEN * HIDDEN;
    const float* src = W + moff;
    bf16_t* dst = Wt + moff;
    const int tx = threadIdx.x & 31;
    const int ty = threadIdx.x >> 5;          // 0..7
    const int n0 = blockIdx.x * 32;
    const int k0 = blockIdx.y * 32;
#pragma unroll
    for (int r = ty; r < 32; r += 8)
        tile[r][tx] = src[(size_t)(k0 + r) * HIDDEN + n0 + tx];
    __syncthreads();
#pragma unroll
    for (int r = ty; r < 32; r += 8)
        dst[(size_t)(n0 + r) * HIDDEN + k0 + tx] = (bf16_t)tile[tx][r];
}

// ---------------- embedding + positional (dual fp32 + bf16 output) ----------------
__global__ __launch_bounds__(256)
void embed_pos_kernel(const int* __restrict__ tokens,
                      const float* __restrict__ emb,
                      float* __restrict__ x, bf16_t* __restrict__ xb) {
    const int row = blockIdx.x;
    const int l   = row & (SEQ - 1);
    const int tok = tokens[row];
    const float* e = emb + (size_t)tok * HIDDEN;
    float*  xr = x  + (size_t)row * HIDDEN;
    bf16_t* xh = xb + (size_t)row * HIDDEN;
    const float LOG1000 = 6.907755278982137f;
    for (int h = threadIdx.x; h < HIDDEN; h += blockDim.x) {
        int i = (h < HIDDEN / 2) ? h : (h - HIDDEN / 2);
        float rate = __expf(-(2.0f * (float)i / (float)HIDDEN) * LOG1000);
        float ang  = (float)l * rate;
        float p    = (h < HIDDEN / 2) ? __sinf(ang) : __cosf(ang);
        float v    = e[h] + p;
        xr[h] = v;
        xh[h] = (bf16_t)v;
    }
}

// ---------------- bf16 WMMA GEMM, operands straight from L2 ----------------
// A   : bf16 [M][1024] row-major (activations)
// Wt  : bf16 [1024 n][1024 k]   (pre-transposed weights)
// out = (acc + bias) * oscale
// MODE 0: Cf fp32 [row][col]
// MODE 1: Cb bf16 [row][col]               (Q, K)
// MODE 2: Cb bf16 vT[b][head][d][key]      (V, pre-transposed for attention)
#define GBM 128
#define GBN 128

template <int MODE>
__global__ __launch_bounds__(256)
void gemm_bf16_kernel(const bf16_t* __restrict__ A,
                      const bf16_t* __restrict__ Wt,
                      const float* __restrict__ bias,
                      float* __restrict__ Cf,
                      bf16_t* __restrict__ Cb,
                      float oscale) {
    const int lane = threadIdx.x & 31;
    const int wave = threadIdx.x >> 5;                  // 0..7
    const int row0 = blockIdx.y * GBM + wave * 16;      // wave's 16-row strip
    const int col0 = blockIdx.x * GBN;

    const int m   = lane & 15;
    const int koA = (lane < 16) ? 0 : 8;    // A frag: elems 0..7 -> K ko..ko+7, 8..15 -> K 16+ko..
    const int koB = (lane < 16) ? 0 : 8;    // B frag: elems 0..7 -> K ko..ko+7, 8..15 -> K 16+ko..
    const int nl  = lane & 15;
    const bf16_t* Arow = A + (size_t)(row0 + m) * HIDDEN;

    v8f acc[8];
#pragma unroll
    for (int t = 0; t < 8; ++t)
#pragma unroll
        for (int j = 0; j < 8; ++j) acc[t][j] = 0.0f;

    for (int k0 = 0; k0 < HIDDEN; k0 += 32) {
        v16bf af = pack16(*(const v8bf*)(Arow + k0 + koA),
                          *(const v8bf*)(Arow + k0 + 16 + koA));
        if (k0 + 32 < HIDDEN)
            __builtin_prefetch((const void*)(Arow + k0 + 32), 0, 3);
#pragma unroll
        for (int nt = 0; nt < 8; ++nt) {
            const bf16_t* wp = Wt + (size_t)(col0 + nt * 16 + nl) * HIDDEN + k0 + koB;
            v16bf bfv = pack16(*(const v8bf*)(wp), *(const v8bf*)(wp + 16));
            acc[nt] = wmma_bf16(af, bfv, acc[nt]);
        }
    }

    // C layout: lanes 0-15 rows 0..7, lanes 16-31 rows 8..15
    const int rbase = row0 + ((lane < 16) ? 0 : 8);
#pragma unroll
    for (int nt = 0; nt < 8; ++nt) {
        const int col = col0 + nt * 16 + nl;
        const float bsc = bias[col];
#pragma unroll
        for (int j = 0; j < 8; ++j) {
            const int row = rbase + j;
            const float val = (acc[nt][j] + bsc) * oscale;
            if (MODE == 0) {
                Cf[(size_t)row * HIDDEN + col] = val;
            } else if (MODE == 1) {
                Cb[(size_t)row * HIDDEN + col] = (bf16_t)val;
            } else {
                const int bi = row >> 10, key = row & (SEQ - 1);
                const int hd = col >> 6,  d   = col & (DHEAD - 1);
                Cb[((size_t)((bi * HEADS + hd) * DHEAD + d) << 10) + key] = (bf16_t)val;
            }
        }
    }
}

// ---------------- flash attention: all operands bf16-native ----------------
// Q is pre-scaled by log2(e)/sqrt(HIDDEN), so softmax runs in exp2 domain.
__global__ __launch_bounds__(128)
void attention_kernel(const bf16_t* __restrict__ Q,    // [4096][1024], pre-scaled
                      const bf16_t* __restrict__ K,    // [4096][1024]
                      const bf16_t* __restrict__ VT,   // [b][head][d][key]
                      float* __restrict__ O) {
    __shared__ __align__(16) bf16_t Ps[4][16 * 32];
    const int lane = threadIdx.x & 31;
    const int wave = threadIdx.x >> 5;         // 0..3, each wave: 16 query rows
    const int qt   = blockIdx.x;
    const int head = blockIdx.y;
    const int b    = blockIdx.z;
    const int hoff = head * DHEAD;
    const int qrow = b * SEQ + qt * 64 + wave * 16;
    const int kro  = b * SEQ;

    const int m     = lane & 15;
    const int koA   = (lane < 16) ? 0 : 8;
    const int dbase = (lane < 16) ? 0 : 8;
    const int nl    = lane & 15;

    // Q fragments: 16 rows x 64 dims -> two 16x32 A fragments (b128 loads)
    v16bf qf[2];
    {
        const bf16_t* qp = Q + (size_t)(qrow + m) * HIDDEN + hoff;
#pragma unroll
        for (int h = 0; h < 2; ++h)
            qf[h] = pack16(*(const v8bf*)(qp + h * 32 + koA),
                           *(const v8bf*)(qp + h * 32 + 16 + koA));
    }

    float mrow[8], lrow[8];
#pragma unroll
    for (int j = 0; j < 8; ++j) { mrow[j] = -1e30f; lrow[j] = 0.0f; }
    v8f oacc[4];
#pragma unroll
    for (int t = 0; t < 4; ++t)
#pragma unroll
        for (int j = 0; j < 8; ++j) oacc[t][j] = 0.0f;

    const bf16_t* vbase = VT + ((size_t)((b * HEADS + head) * DHEAD) << 10);

    for (int kc = 0; kc < SEQ; kc += 32) {
        // ---- S = Qs * K^T for 32 keys (already in log2 domain)
        v8f s[2];
#pragma unroll
        for (int kt = 0; kt < 2; ++kt) {
            v8f sa;
#pragma unroll
            for (int j = 0; j < 8; ++j) sa[j] = 0.0f;
            const bf16_t* kp = K + (size_t)(kro + kc + kt * 16 + nl) * HIDDEN + hoff;
#pragma unroll
            for (int h = 0; h < 2; ++h) {
                v16bf kf = pack16(*(const v8bf*)(kp + h * 32 + dbase),
                                  *(const v8bf*)(kp + h * 32 + dbase + 16));
                sa = wmma_bf16(qf[h], kf, sa);
            }
            s[kt] = sa;
        }

        // ---- online softmax (lanes 0-15 <-> rows 0..7, lanes 16-31 <-> rows 8..15)
        bf16_t* ps = Ps[wave];
#pragma unroll
        for (int j = 0; j < 8; ++j) {
            float s0 = s[0][j];
            float s1 = s[1][j];
            float t = fmaxf(s0, s1);
            t = fmaxf(t, __shfl_xor(t, 1, 32));
            t = fmaxf(t, __shfl_xor(t, 2, 32));
            t = fmaxf(t, __shfl_xor(t, 4, 32));
            t = fmaxf(t, __shfl_xor(t, 8, 32));
            float mnew = fmaxf(mrow[j], t);
            float p0 = exp2f(s0 - mnew);
            float p1 = exp2f(s1 - mnew);
            float rs = p0 + p1;
            rs += __shfl_xor(rs, 1, 32);
            rs += __shfl_xor(rs, 2, 32);
            rs += __shfl_xor(rs, 4, 32);
            rs += __shfl_xor(rs, 8, 32);
            float alpha = exp2f(mrow[j] - mnew);
            lrow[j] = lrow[j] * alpha + rs;
            mrow[j] = mnew;
#pragma unroll
            for (int nt = 0; nt < 4; ++nt) oacc[nt][j] *= alpha;
            const int r = (lane < 16) ? j : (8 + j);
            ps[r * 32 + nl]      = (bf16_t)p0;
            ps[r * 32 + 16 + nl] = (bf16_t)p1;
        }

        // ---- reload P as A fragment (LDS in-order per wave; wave-private tile)
        const bf16_t* pp = ps + m * 32;
        v16bf pf = pack16(*(const v8bf*)(pp + koA),
                          *(const v8bf*)(pp + 16 + koA));

        // ---- O += P(16x32) * V(32x64), V pre-transposed: contiguous b128 loads
#pragma unroll
        for (int nt = 0; nt < 4; ++nt) {
            const bf16_t* vp = vbase + ((size_t)(nt * 16 + nl) << 10) + kc + dbase;
            v16bf vf = pack16(*(const v8bf*)(vp), *(const v8bf*)(vp + 16));
            oacc[nt] = wmma_bf16(pf, vf, oacc[nt]);
        }
    }

    // ---- normalize + store fp32
    const int rbase = qrow + ((lane < 16) ? 0 : 8);
#pragma unroll
    for (int nt = 0; nt < 4; ++nt)
#pragma unroll
        for (int j = 0; j < 8; ++j)
            O[(size_t)(rbase + j) * HIDDEN + hoff + nt * 16 + nl] =
                oacc[nt][j] / lrow[j];
}

// ---------------- fused residual add + LayerNorm (dual fp32 + bf16 out) ----------------
__global__ __launch_bounds__(256)
void add_ln_kernel(const float* __restrict__ A, const float* __restrict__ B,
                   const float* __restrict__ gamma, const float* __restrict__ beta,
                   float* __restrict__ out, bf16_t* __restrict__ outb) {
    __shared__ float red[16];
    const int row = blockIdx.x;
    const float* a = A + (size_t)row * HIDDEN;
    const float* b = B + (size_t)row * HIDDEN;
    float*  o  = out  + (size_t)row * HIDDEN;
    bf16_t* ob = outb + (size_t)row * HIDDEN;

    float v[4];
    float s = 0.0f, ss = 0.0f;
#pragma unroll
    for (int j = 0; j < 4; ++j) {
        int h = threadIdx.x + j * 256;
        float t = a[h] + b[h];
        v[j] = t;
        s += t; ss += t * t;
    }
#pragma unroll
    for (int mk = 1; mk < 32; mk <<= 1) {
        s  += __shfl_xor(s, mk, 32);
        ss += __shfl_xor(ss, mk, 32);
    }
    int wave = threadIdx.x >> 5, lane = threadIdx.x & 31;
    if (lane == 0) { red[wave] = s; red[8 + wave] = ss; }
    __syncthreads();
    s = 0.0f; ss = 0.0f;
#pragma unroll
    for (int w = 0; w < 8; ++w) { s += red[w]; ss += red[8 + w]; }
    float mu   = s * (1.0f / HIDDEN);
    float var  = ss * (1.0f / HIDDEN) - mu * mu;
    float rstd = rsqrtf(var + EPS_LN);
#pragma unroll
    for (int j = 0; j < 4; ++j) {
        int h = threadIdx.x + j * 256;
        float t = (v[j] - mu) * rstd * gamma[h] + beta[h];
        o[h]  = t;
        ob[h] = (bf16_t)t;
    }
}

// ---------------- host orchestration ----------------
extern "C" void kernel_launch(void* const* d_in, const int* in_sizes, int n_in,
                              void* d_out, int out_size, void* d_ws, size_t ws_size,
                              hipStream_t stream) {
    (void)in_sizes; (void)n_in; (void)out_size; (void)ws_size;

    const int*   tokens = (const int*)  d_in[0];
    const float* emb    = (const float*)d_in[1];
    const float* wq     = (const float*)d_in[2];
    const float* bq     = (const float*)d_in[3];
    const float* wk     = (const float*)d_in[4];
    const float* bk     = (const float*)d_in[5];
    const float* wv     = (const float*)d_in[6];
    const float* bv     = (const float*)d_in[7];
    const float* wd     = (const float*)d_in[8];
    const float* bd     = (const float*)d_in[9];
    const float* g1     = (const float*)d_in[10];
    const float* b1     = (const float*)d_in[11];
    const float* g2     = (const float*)d_in[12];
    const float* b2     = (const float*)d_in[13];

    const size_t T  = (size_t)ROWS * HIDDEN;     // 4M elements
    const size_t WM = (size_t)HIDDEN * HIDDEN;   // 1M elements per matrix

    char* p = (char*)d_ws;
    float*  x    = (float*)p;  p += T * 4;
    float*  x1   = (float*)p;  p += T * 4;
    float*  o    = (float*)p;  p += T * 4;
    float*  y    = (float*)p;  p += T * 4;
    bf16_t* xb   = (bf16_t*)p; p += T * 2;
    bf16_t* x1b  = (bf16_t*)p; p += T * 2;
    bf16_t* qb   = (bf16_t*)p; p += T * 2;
    bf16_t* kb   = (bf16_t*)p; p += T * 2;
    bf16_t* vtb  = (bf16_t*)p; p += T * 2;
    bf16_t* wtq  = (bf16_t*)p; p += WM * NBLOCKS * 2;
    bf16_t* wtk  = (bf16_t*)p; p += WM * NBLOCKS * 2;
    bf16_t* wtv  = (bf16_t*)p; p += WM * NBLOCKS * 2;
    bf16_t* wtd  = (bf16_t*)p; p += WM * 2;

    // one-time per launch: weights -> bf16, transposed to [n][k]
    dim3 tgridQ(32, 32, NBLOCKS);
    transpose_cvt_kernel<<<tgridQ, 256, 0, stream>>>(wq, wtq);
    transpose_cvt_kernel<<<tgridQ, 256, 0, stream>>>(wk, wtk);
    transpose_cvt_kernel<<<tgridQ, 256, 0, stream>>>(wv, wtv);
    transpose_cvt_kernel<<<dim3(32, 32, 1), 256, 0, stream>>>(wd, wtd);

    embed_pos_kernel<<<ROWS, 256, 0, stream>>>(tokens, emb, x, xb);

    dim3 ggrid(HIDDEN / GBN, ROWS / GBM);      // (8, 32)
    dim3 agrid(SEQ / 64, HEADS, BATCH);        // (16, 16, 4)

    for (int i = 0; i < NBLOCKS; ++i) {
        const bf16_t* wqi = wtq + (size_t)i * WM;
        const bf16_t* wki = wtk + (size_t)i * WM;
        const bf16_t* wvi = wtv + (size_t)i * WM;
        const float*  bqi = bq + (size_t)i * HIDDEN;
        const float*  bki = bk + (size_t)i * HIDDEN;
        const float*  bvi = bv + (size_t)i * HIDDEN;

        // Q carries the softmax scale (in log2 domain); K/V are unscaled.
        gemm_bf16_kernel<1><<<ggrid, 256, 0, stream>>>(xb, wqi, bqi, nullptr, qb, QSCALE_LOG2E);
        gemm_bf16_kernel<1><<<ggrid, 256, 0, stream>>>(xb, wki, bki, nullptr, kb, 1.0f);
        gemm_bf16_kernel<2><<<ggrid, 256, 0, stream>>>(xb, wvi, bvi, nullptr, vtb, 1.0f);
        attention_kernel<<<agrid, 128, 0, stream>>>(qb, kb, vtb, o);
        add_ln_kernel<<<ROWS, 256, 0, stream>>>(x, o, g1, b1, x1, x1b);
        gemm_bf16_kernel<0><<<ggrid, 256, 0, stream>>>(x1b, wtd, bd, y, nullptr, 1.0f);
        float* xo = (i == NBLOCKS - 1) ? (float*)d_out : x;
        add_ln_kernel<<<ROWS, 256, 0, stream>>>(x1, y, g2, b2, xo, xb);
    }
}
